// DiscreteBipartiteFlow_7868380086669
// MI455X (gfx1250) — compile-verified, matrix operand order checked by reading, not run on hardware
//
#include <hip/hip_runtime.h>
#include <hip/hip_bf16.h>

// DiscreteBipartiteFlow forward for MI455X (gfx1250, wave32, WMMA).
// net = relu(x0@W1+b1)@W2+b2 via v_wmma_f32_16x16x32_f16; weight panels are
// staged per-workgroup in LDS with gfx1250 async global->LDS copies
// (ASYNCcnt path) to kill redundant per-wave L2 streaming. Discrete head
// (straight-through argmax one-hots + modular one-hot multiply/add) reduced
// to per-row argmax + index arithmetic: z1 = onehot((l* + a1*s*) mod 128),
// zero row when s*==0 (scale index 0 excluded in the reference).

// Flip to 0 if the async-to-LDS mnemonic is rejected by the assembler.
#define USE_ASYNC_LDS 1

typedef __attribute__((ext_vector_type(16))) _Float16 v16h;
typedef __attribute__((ext_vector_type(8)))  _Float16 v8h;
typedef __attribute__((ext_vector_type(8)))  float    v8f;

#define B_  8192
#define V_  128
#define H_  512

static __device__ __forceinline__ v16h mk_frag(v8h lo, v8h hi) {
  v16h r;
#pragma unroll
  for (int e = 0; e < 8; ++e) { r[e] = lo[e]; r[8 + e] = hi[e]; }
  return r;
}

// 16B global -> LDS copy. Async variant uses the gfx1250 ASYNC path
// (GLOBAL_LOAD_ASYNC_TO_LDS_B128, GV addressing: dsaddr from VGPR, 64b vaddr).
static __device__ __forceinline__ void copy16_g2lds(void* lds_dst, const void* gsrc) {
#if USE_ASYNC_LDS
  unsigned lds_off = (unsigned)(uintptr_t)lds_dst;       // low 32b of flat = LDS offset
  unsigned long long ga = (unsigned long long)(uintptr_t)gsrc;
  asm volatile("global_load_async_to_lds_b128 %0, %1, off"
               :: "v"(lds_off), "v"(ga) : "memory");
#else
  *(v8h*)lds_dst = *(const v8h*)gsrc;
#endif
}

static __device__ __forceinline__ void wait_g2lds() {
#if USE_ASYNC_LDS
  asm volatile("s_wait_asynccnt 0" ::: "memory");
#endif
}

// Butterfly argmax across the 16 lanes of one half-wave (xor masks < 16 keep
// lanes inside their half, matching the WMMA C/D half-layout). Ties -> lower index.
static __device__ __forceinline__ void argmax_half16(float& v, int& i) {
#pragma unroll
  for (int off = 1; off < 16; off <<= 1) {
    float ov = __shfl_xor(v, off, 32);
    int   oi = __shfl_xor(i, off, 32);
    if (ov > v || (ov == v && oi < i)) { v = ov; i = oi; }
  }
}

// ---------------- Kernel 0: weights -> f16, column-major -------------------
// W1 (128x512) -> W1T (512x128), W2 (512x256) -> W2T (256x512) so every WMMA
// B-fragment load is a contiguous 16B run per lane.
__global__ __launch_bounds__(256) void pack_weights(const float* __restrict__ W1,
                                                    const float* __restrict__ W2,
                                                    _Float16* __restrict__ W1T,
                                                    _Float16* __restrict__ W2T) {
  int i = blockIdx.x * blockDim.x + threadIdx.x;   // 0 .. 196607 exactly
  if (i < H_ * V_) {                               // 512*128
    int n = i >> 7, k = i & 127;
    W1T[i] = (_Float16)W1[k * H_ + n];
  } else {
    int j = i - H_ * V_;                           // 0 .. 256*512-1
    int n = j >> 9, k = j & 511;
    W2T[j] = (_Float16)W2[k * (2 * V_) + n];
  }
}

// ---------------- Kernel 1: Hbuf = relu(x0 @ W1 + b1) as f16 ----------------
// 64 blocks x 8 waves; wave owns rows [m0, m0+16). W1T staged in LDS in 4
// N-chunks of 128 cols x K=128 (rows padded to 272B: lane lr hits bank 4*lr).
#define G1_STRIDE 272   // bytes per LDS row (128 f16 = 256B + 16B pad)
__global__ __launch_bounds__(256) void gemm1_relu(const float* __restrict__ inp,
                                                  const float* __restrict__ b1,
                                                  const _Float16* __restrict__ W1T,
                                                  _Float16* __restrict__ Hbuf) {
  __shared__ __align__(16) char smem[128 * G1_STRIDE];   // 34 KB

  const int tid  = threadIdx.x;
  const int lane = tid & 31;
  const int wave = tid >> 5;
  const int lr   = lane & 15;
  const int hf   = lane >> 4;                 // half-wave select
  const int m0   = blockIdx.x * 128 + wave * 16;
  const int m    = m0 + lr;

  // A fragments: CDNA5 16-bit A layout -> lane half picks K 0-7/16-23 vs 8-15/24-31.
  const float* arow = inp + (size_t)m * (2 * V_);   // x0 = cols [0,128)
  v16h afrag[4];
#pragma unroll
  for (int ks = 0; ks < 4; ++ks) {
    const int k0 = ks * 32 + hf * 8;
    const int k1 = ks * 32 + 16 + hf * 8;
    float4 f0 = *(const float4*)(arow + k0);
    float4 f1 = *(const float4*)(arow + k0 + 4);
    float4 f2 = *(const float4*)(arow + k1);
    float4 f3 = *(const float4*)(arow + k1 + 4);
    v16h a;
    a[0]=(_Float16)f0.x;  a[1]=(_Float16)f0.y;  a[2]=(_Float16)f0.z;  a[3]=(_Float16)f0.w;
    a[4]=(_Float16)f1.x;  a[5]=(_Float16)f1.y;  a[6]=(_Float16)f1.z;  a[7]=(_Float16)f1.w;
    a[8]=(_Float16)f2.x;  a[9]=(_Float16)f2.y;  a[10]=(_Float16)f2.z; a[11]=(_Float16)f2.w;
    a[12]=(_Float16)f3.x; a[13]=(_Float16)f3.y; a[14]=(_Float16)f3.z; a[15]=(_Float16)f3.w;
    afrag[ks] = a;
  }

  for (int nc = 0; nc < 4; ++nc) {                 // N chunks of 128 columns
    // Stage chunk: 128 rows x 256B; thread t copies row t/2, 8 of 16 segments.
    {
      const int row = tid >> 1;
      const int sb  = (tid & 1) * 8;
      const _Float16* gsrc = W1T + (size_t)(nc * 128 + row) * V_;
#pragma unroll
      for (int j = 0; j < 8; ++j) {
        const int seg = sb + j;
        copy16_g2lds(smem + row * G1_STRIDE + seg * 16, gsrc + seg * 8);
      }
      wait_g2lds();
    }
    __syncthreads();

#pragma unroll
    for (int ntl = 0; ntl < 8; ++ntl) {            // 8 N-tiles per chunk
      const int n = nc * 128 + ntl * 16 + lr;
      const char* brow = smem + (ntl * 16 + lr) * G1_STRIDE;
      v8f acc = {};
#pragma unroll
      for (int ks = 0; ks < 4; ++ks) {
        v8h lo = *(const v8h*)(brow + ks * 64 + hf * 16);
        v8h hi = *(const v8h*)(brow + ks * 64 + 32 + hf * 16);
        acc = __builtin_amdgcn_wmma_f32_16x16x32_f16(false, afrag[ks], false,
                                                     mk_frag(lo, hi),
                                                     (short)0, acc, false, false);
      }
      const float bias = b1[n];
#pragma unroll
      for (int r = 0; r < 8; ++r) {                // D row = r + 8*hf
        float v = acc[r] + bias;
        v = v > 0.0f ? v : 0.0f;
        Hbuf[(size_t)(m0 + hf * 8 + r) * H_ + n] = (_Float16)v;
      }
    }
    __syncthreads();
  }
}

// ------- Kernel 2: net = Hbuf @ W2 + b2, fused argmax + discrete flow -------
// Wave holds all 16 accumulators (128 VGPRs). W2T staged in LDS in 8 K-chunks
// of 256 cols x K=64 (rows padded to 144B: lane lr hits bank (36*lr)%64, all
// distinct). H A-fragments are per-wave-unique, read straight from global.
#define G2_STRIDE 144   // bytes per LDS row (64 f16 = 128B + 16B pad)
__global__ __launch_bounds__(256) void gemm2_flow(const float* __restrict__ inp,
                                                  const float* __restrict__ b2,
                                                  const _Float16* __restrict__ Hbuf,
                                                  const _Float16* __restrict__ W2T,
                                                  float* __restrict__ out) {
  __shared__ __align__(16) char smem[256 * G2_STRIDE];   // 36 KB

  const int tid  = threadIdx.x;
  const int lane = tid & 31;
  const int wave = tid >> 5;
  const int lr   = lane & 15;
  const int hf   = lane >> 4;
  const int m0   = blockIdx.x * 128 + wave * 16;
  const int m    = m0 + lr;

  float b2c[16];
#pragma unroll
  for (int n = 0; n < 16; ++n) b2c[n] = b2[n * 16 + lr];

  v8f acc[16];
  {
    v8f z = {};
#pragma unroll
    for (int n = 0; n < 16; ++n) acc[n] = z;
  }

  const _Float16* hrow = Hbuf + (size_t)m * H_;
  for (int kc = 0; kc < 8; ++kc) {                 // K chunks of 64
    // Stage chunk: 256 rows x 128B; thread t copies its whole row (8 segs).
    {
      const _Float16* gsrc = W2T + (size_t)tid * H_ + kc * 64;
#pragma unroll
      for (int j = 0; j < 8; ++j)
        copy16_g2lds(smem + tid * G2_STRIDE + j * 16, gsrc + j * 8);
      wait_g2lds();
    }
    __syncthreads();

#pragma unroll
    for (int ksl = 0; ksl < 2; ++ksl) {            // 2 k-steps of 32 per chunk
      const int kb = kc * 64 + ksl * 32;
      v8h alo = *(const v8h*)(hrow + kb + hf * 8);
      v8h ahi = *(const v8h*)(hrow + kb + 16 + hf * 8);
      const v16h a = mk_frag(alo, ahi);
#pragma unroll
      for (int n = 0; n < 16; ++n) {
        const char* brow = smem + (n * 16 + lr) * G2_STRIDE;
        v8h blo = *(const v8h*)(brow + ksl * 64 + hf * 16);
        v8h bhi = *(const v8h*)(brow + ksl * 64 + 32 + hf * 16);
        acc[n] = __builtin_amdgcn_wmma_f32_16x16x32_f16(false, a, false,
                                                        mk_frag(blo, bhi),
                                                        (short)0, acc[n], false, false);
      }
    }
    __syncthreads();
  }

  // Epilogue: per accumulator row r, lanes<16 own row m0+r, lanes>=16 own m0+8+r.
#pragma unroll
  for (int r = 0; r < 8; ++r) {
    const int row = m0 + hf * 8 + r;

    // l* = argmax(loc): N-tiles 0..7 cover columns [0,128)
    float lv = -__builtin_inff(); int li = 0;
#pragma unroll
    for (int n = 0; n < 8; ++n) {
      float v = acc[n][r] + b2c[n];
      int idx = n * 16 + lr;
      if (v > lv || (v == lv && idx < li)) { lv = v; li = idx; }
    }
    argmax_half16(lv, li);

    // s* = argmax(scale): N-tiles 8..15 cover columns [128,256)
    float sv = -__builtin_inff(); int si = 0;
#pragma unroll
    for (int n = 8; n < 16; ++n) {
      float v = acc[n][r] + b2c[n];
      int idx = (n - 8) * 16 + lr;
      if (v > sv || (v == sv && idx < si)) { sv = v; si = idx; }
    }
    argmax_half16(sv, si);

    // a1 = argmax of one-hot x1 row
    const float* x1row = inp + (size_t)row * (2 * V_) + V_;
    float av = -__builtin_inff(); int ai = 0;
#pragma unroll
    for (int j = 0; j < 8; ++j) {
      int idx = lr * 8 + j;
      float v = x1row[idx];
      if (v > av || (v == av && idx < ai)) { av = v; ai = idx; }
    }
    argmax_half16(av, ai);

    // one_hot_multiply then circular-convolution add, in index space.
    const int tgt = (li + ai * si) & (V_ - 1);

    const float* irow = inp + (size_t)row * (2 * V_);
    float* orow = out + (size_t)row * (2 * V_);
#pragma unroll
    for (int j = 0; j < 16; ++j) {                 // 16 lanes x 16 cols = 256
      int c = lr * 16 + j;
      float v;
      if (c < V_) v = irow[c];                                    // x0 passthrough
      else        v = (si >= 1 && (c - V_) == tgt) ? 1.0f : 0.0f; // z1 one-hot (0 if s*==0)
      orow[c] = v;
    }
  }
}

extern "C" void kernel_launch(void* const* d_in, const int* in_sizes, int n_in,
                              void* d_out, int out_size, void* d_ws, size_t ws_size,
                              hipStream_t stream) {
  (void)in_sizes; (void)n_in; (void)out_size; (void)ws_size;
  const float* inp = (const float*)d_in[0];
  const float* W1  = (const float*)d_in[1];
  const float* b1  = (const float*)d_in[2];
  const float* W2  = (const float*)d_in[3];
  const float* b2  = (const float*)d_in[4];
  float* out = (float*)d_out;

  // Workspace: Hbuf f16 8 MB | W1T f16 128 KB | W2T f16 256 KB  (~8.4 MB)
  _Float16* Hbuf = (_Float16*)d_ws;
  _Float16* W1T  = (_Float16*)((char*)d_ws + (size_t)B_ * H_ * sizeof(_Float16));
  _Float16* W2T  = W1T + (size_t)H_ * V_;

  pack_weights<<<dim3((H_ * V_ + 2 * V_ * H_) / 256), dim3(256), 0, stream>>>(W1, W2, W1T, W2T);
  gemm1_relu<<<dim3(B_ / 128), dim3(256), 0, stream>>>(inp, b1, W1T, Hbuf);
  gemm2_flow<<<dim3(B_ / 128), dim3(256), 0, stream>>>(inp, b2, Hbuf, W2T, out);
}
